// DCNv2_9036611191065
// MI455X (gfx1250) — matrix operand order repaired and checked by reading, not compiled, not run on hardware
//
#include <hip/hip_runtime.h>
#include <math.h>

typedef __attribute__((ext_vector_type(16))) __bf16 v16bf;
typedef __attribute__((ext_vector_type(8)))  float  v8f;

#define B_   2
#define C_   256
#define H_   64
#define W_   64
#define HW_  4096
#define TAPS 9
#define NCC  8      // 256 channels / 32-chunk
#define NQ   (TAPS * NCC)   // 72 K-chunks

// ---------- helpers ----------
__device__ __forceinline__ unsigned short f2bf(float f) {
    unsigned u = __float_as_uint(f);
    unsigned r = u + 0x7FFFu + ((u >> 16) & 1u);   // round-to-nearest-even
    return (unsigned short)(r >> 16);
}
__device__ __forceinline__ unsigned pack2(float lo, float hi) {
    return (unsigned)f2bf(lo) | ((unsigned)f2bf(hi) << 16);
}
__device__ __forceinline__ v8f wmma_bf16(v16bf a, v16bf b, v8f c) {
    return __builtin_amdgcn_wmma_f32_16x16x32_bf16(
        false, a, false, b, (short)0, c, false, false);
}

// A-fragment (16x32 bf16) element mapping, per CDNA5 ISA 7.12.2:
//   lane L (M = L&15), dword v, half h:
//   K = 2*(v&3) + h + (v>=4 ? 16 : 0) + (L>=16 ? 8 : 0)
__device__ __forceinline__ int afrag_k(int v, int h, int lane) {
    return 2 * (v & 3) + h + ((v >= 4) ? 16 : 0) + ((lane >= 16) ? 8 : 0);
}

// ---------- weight repack: w_dcn [256][256][3][3] f32 -> bf16 A-frags ----------
// layout: [tap][cc][mtile(16)][lane(32)][dword(8)]
__global__ void repack_dcn(const float* __restrict__ w, unsigned* __restrict__ dst) {
    unsigned id = blockIdx.x * 256u + threadIdx.x;      // 294912 total
    int v     = id & 7;
    int lane  = (id >> 3) & 31;
    int mtile = (id >> 8) & 15;
    int cc    = (id >> 12) & 7;
    int tap   = id >> 15;
    int o  = mtile * 16 + (lane & 15);
    int c0 = cc * 32 + afrag_k(v, 0, lane);
    int c1 = cc * 32 + afrag_k(v, 1, lane);
    float f0 = w[(o * C_ + c0) * TAPS + tap];
    float f1 = w[(o * C_ + c1) * TAPS + tap];
    dst[id] = pack2(f0, f1);
}

// ---------- weight repack: w_offset_mask [27][256][3][3] f32 (pad M->32) ----------
// layout: [tap][cc][mtile(2)][lane(32)][dword(8)]
__global__ void repack_off(const float* __restrict__ w, unsigned* __restrict__ dst) {
    unsigned id = blockIdx.x * 256u + threadIdx.x;      // 36864 total
    int v     = id & 7;
    int lane  = (id >> 3) & 31;
    int mtile = (id >> 8) & 1;
    int cc    = (id >> 9) & 7;
    int tap   = id >> 12;
    int o  = mtile * 16 + (lane & 15);
    int c0 = cc * 32 + afrag_k(v, 0, lane);
    int c1 = cc * 32 + afrag_k(v, 1, lane);
    float f0 = (o < 27) ? w[(o * C_ + c0) * TAPS + tap] : 0.0f;
    float f1 = (o < 27) ? w[(o * C_ + c1) * TAPS + tap] : 0.0f;
    dst[id] = pack2(f0, f1);
}

// ---------- offset/mask conv (implicit GEMM, M=32, N=64/block) ----------
// grid = B*H blocks of 256 threads. Writes dy/dx (raw) and sigmoid(mask) scratch.
__global__ void __launch_bounds__(256)
conv_offset(const float* __restrict__ of, const float* __restrict__ bias,
            const unsigned* __restrict__ wsA,
            float* __restrict__ dyA, float* __restrict__ dxA, float* __restrict__ mkA)
{
    __shared__ __attribute__((aligned(32))) unsigned ldsB[2][4 * 32 * 8];  // 2 x 4KB

    const int bid  = blockIdx.x;
    const int b    = bid >> 6;
    const int y    = bid & 63;
    const int tid  = threadIdx.x;
    const int wave = tid >> 5;
    const int lane = tid & 31;
    const int mtile = wave & 1;     // M sub-tile (2)
    const int ntile = wave >> 1;    // N sub-tile (4)

    // gather role
    const int p     = tid & 63;            // pixel (x) this thread gathers
    const int cbase = (tid >> 6) * 8;      // 8 consecutive channels in chunk
    const unsigned laneW  = (unsigned)((p & 15) + ((cbase & 8) ? 16 : 0));
    const unsigned ldsOff = (unsigned)((p >> 4) * 256 + laneW * 8 + ((cbase >> 4) * 4));

    // accumulator init with bias (C/D layout: VGPR j -> M = j + (lane>=16?8:0))
    v8f acc;
    #pragma unroll
    for (int j = 0; j < 8; ++j) {
        int ch = mtile * 16 + j + ((lane >= 16) ? 8 : 0);
        acc[j] = (ch < 27) ? bias[ch] : 0.0f;
    }

    const float* ofb = of + (size_t)b * C_ * HW_;

    // branch-free gather of one 32-channel chunk of shifted-window im2col
    int   pidx = 0;
    float okf  = 0.0f;
    auto loadParams = [&](int tap) {
        const int ky = tap / 3, kx = tap % 3;
        const int iy = y + ky - 1;
        const int ix = p + kx - 1;
        okf  = ((iy >= 0) && (iy < H_) && (ix >= 0) && (ix < W_)) ? 1.0f : 0.0f;
        const int ciy = min(max(iy, 0), H_ - 1);
        const int cix = min(max(ix, 0), W_ - 1);
        pidx = ciy * W_ + cix;
    };
    auto gatherChunk = [&](int q, unsigned* buf) {
        const int cc = q & 7;
        const float* plane = ofb + (size_t)(cc * 32 + cbase) * HW_;
        float vals[8];
        #pragma unroll
        for (int i = 0; i < 8; ++i)
            vals[i] = plane[(size_t)i * HW_ + pidx] * okf;
        #pragma unroll
        for (int i = 0; i < 4; ++i)
            buf[ldsOff + i] = pack2(vals[2 * i], vals[2 * i + 1]);
    };

    loadParams(0);
    gatherChunk(0, ldsB[0]);
    for (int q = 0; q < NQ; ++q) {
        __syncthreads();
        if (q + 1 < NQ) {
            if (((q + 1) & 7) == 0) loadParams((q + 1) >> 3);
            gatherChunk(q + 1, ldsB[(q + 1) & 1]);
        }
        const int tap = q >> 3, cc = q & 7;
        v16bf aF = *reinterpret_cast<const v16bf*>(
            wsA + (size_t)(((tap * NCC + cc) * 2 + mtile) * 32 + lane) * 8);
        v16bf bF = *reinterpret_cast<const v16bf*>(&ldsB[q & 1][(ntile * 32 + lane) * 8]);
        acc = wmma_bf16(aF, bF, acc);
    }

    // epilogue: split channels -> dy/dx/mask
    const int xo = ntile * 16 + (lane & 15);
    #pragma unroll
    for (int j = 0; j < 8; ++j) {
        int ch = mtile * 16 + j + ((lane >= 16) ? 8 : 0);
        float v = acc[j];
        if (ch < 18) {
            int tap = ch >> 1;
            size_t idx = ((size_t)(b * TAPS + tap) << 12) + y * W_ + xo;
            if (ch & 1) dxA[idx] = v; else dyA[idx] = v;
        } else if (ch < 27) {
            int tap = ch - 18;
            size_t idx = ((size_t)(b * TAPS + tap) << 12) + y * W_ + xo;
            mkA[idx] = 1.0f / (1.0f + __expf(-v));
        }
    }
}

// ---------- main fused deformable conv (implicit GEMM, M=256, N=64/block) ----------
__global__ void __launch_bounds__(256)
dcn_main(const float* __restrict__ x, const float* __restrict__ bias,
         const unsigned* __restrict__ wsA,
         const float* __restrict__ dyA, const float* __restrict__ dxA,
         const float* __restrict__ mkA, float* __restrict__ out)
{
    __shared__ __attribute__((aligned(32))) unsigned ldsB[2][4 * 32 * 8];  // 2 x 4KB

    const int bid  = blockIdx.x;
    const int b    = bid >> 6;
    const int y    = bid & 63;
    const int tid  = threadIdx.x;
    const int wave = tid >> 5;
    const int lane = tid & 31;

    // gather role
    const int p     = tid & 63;
    const int cbase = (tid >> 6) * 8;
    const unsigned laneW  = (unsigned)((p & 15) + ((cbase & 8) ? 16 : 0));
    const unsigned ldsOff = (unsigned)((p >> 4) * 256 + laneW * 8 + ((cbase >> 4) * 4));

    // 8 accumulator tiles per wave: mt in {0,1} (M = (wave*2+mt)*16), nt in 0..3
    v8f acc[8];
    #pragma unroll
    for (int mt = 0; mt < 2; ++mt)
        #pragma unroll
        for (int nt = 0; nt < 4; ++nt) {
            v8f a;
            #pragma unroll
            for (int j = 0; j < 8; ++j) {
                int o = (wave * 2 + mt) * 16 + j + ((lane >= 16) ? 8 : 0);
                a[j] = bias[o];
            }
            acc[mt * 4 + nt] = a;
        }

    const float* xb = x + (size_t)b * C_ * HW_;

    // per-(pixel,tap) bilinear sampling state (registers)
    int   i00 = 0, i01 = 0, i10 = 0, i11 = 0;
    float W00 = 0.0f, W01 = 0.0f, W10 = 0.0f, W11 = 0.0f;

    auto loadParams = [&](int tap) {
        const int ky = tap / 3, kx = tap % 3;
        size_t sidx = ((size_t)(b * TAPS + tap) << 12) + y * W_ + p;
        float dy = dyA[sidx], dx = dxA[sidx], m = mkA[sidx];

        float ys = (float)(y + ky - 1) + dy;
        float xs = (float)(p + kx - 1) + dx;
        float y0f = floorf(ys), x0f = floorf(xs);
        float ly = ys - y0f, lx = xs - x0f;
        int y0 = (int)y0f, x0 = (int)x0f;
        int y1 = y0 + 1, x1 = x0 + 1;

        int cy0 = min(max(y0, 0), H_ - 1), cy1 = min(max(y1, 0), H_ - 1);
        int cx0 = min(max(x0, 0), W_ - 1), cx1 = min(max(x1, 0), W_ - 1);
        float vy0 = (y0 >= 0 && y0 < H_) ? 1.0f : 0.0f;
        float vy1 = (y1 >= 0 && y1 < H_) ? 1.0f : 0.0f;
        float vx0 = (x0 >= 0 && x0 < W_) ? 1.0f : 0.0f;
        float vx1 = (x1 >= 0 && x1 < W_) ? 1.0f : 0.0f;

        i00 = cy0 * W_ + cx0; i01 = cy0 * W_ + cx1;
        i10 = cy1 * W_ + cx0; i11 = cy1 * W_ + cx1;
        W00 = (1.0f - ly) * (1.0f - lx) * vy0 * vx0 * m;
        W01 = (1.0f - ly) * lx          * vy0 * vx1 * m;
        W10 = ly          * (1.0f - lx) * vy1 * vx0 * m;
        W11 = ly          * lx          * vy1 * vx1 * m;
    };
    auto gatherChunk = [&](int q, unsigned* buf) {
        const int cc = q & 7;
        const float* plane = xb + (size_t)(cc * 32 + cbase) * HW_;
        float vals[8];
        #pragma unroll
        for (int i = 0; i < 8; ++i) {
            const float* pl = plane + (size_t)i * HW_;
            vals[i] = W00 * pl[i00] + W01 * pl[i01] + W10 * pl[i10] + W11 * pl[i11];
        }
        #pragma unroll
        for (int i = 0; i < 4; ++i)
            buf[ldsOff + i] = pack2(vals[2 * i], vals[2 * i + 1]);
    };

    loadParams(0);
    gatherChunk(0, ldsB[0]);
    for (int q = 0; q < NQ; ++q) {
        __syncthreads();
        // gather-ahead: fill the other buffer while WMMAs consume this one
        if (q + 1 < NQ) {
            if (((q + 1) & 7) == 0) loadParams((q + 1) >> 3);
            gatherChunk(q + 1, ldsB[(q + 1) & 1]);
        }
        const int tap = q >> 3, cc = q & 7;
        const unsigned* buf = ldsB[q & 1];
        v16bf aF[2], bF[4];
        #pragma unroll
        for (int mt = 0; mt < 2; ++mt)
            aF[mt] = *reinterpret_cast<const v16bf*>(
                wsA + (size_t)(((tap * NCC + cc) * 16 + wave * 2 + mt) * 32 + lane) * 8);
        #pragma unroll
        for (int nt = 0; nt < 4; ++nt)
            bF[nt] = *reinterpret_cast<const v16bf*>(&buf[(nt * 32 + lane) * 8]);
        #pragma unroll
        for (int mt = 0; mt < 2; ++mt)
            #pragma unroll
            for (int nt = 0; nt < 4; ++nt)
                acc[mt * 4 + nt] = wmma_bf16(aF[mt], bF[nt], acc[mt * 4 + nt]);
    }

    // epilogue: store NCHW fp32 output
    #pragma unroll
    for (int mt = 0; mt < 2; ++mt)
        #pragma unroll
        for (int nt = 0; nt < 4; ++nt) {
            int xo = nt * 16 + (lane & 15);
            #pragma unroll
            for (int j = 0; j < 8; ++j) {
                int o = (wave * 2 + mt) * 16 + j + ((lane >= 16) ? 8 : 0);
                out[((size_t)(b * C_ + o) << 12) + y * W_ + xo] = acc[mt * 4 + nt][j];
            }
        }
}

// ---------- launch ----------
extern "C" void kernel_launch(void* const* d_in, const int* in_sizes, int n_in,
                              void* d_out, int out_size, void* d_ws, size_t ws_size,
                              hipStream_t stream) {
    const float* x      = (const float*)d_in[0];
    const float* of     = (const float*)d_in[1];
    const float* w_om   = (const float*)d_in[2];
    const float* b_om   = (const float*)d_in[3];
    const float* w_dcn  = (const float*)d_in[4];
    const float* b_dcn  = (const float*)d_in[5];
    float* out = (float*)d_out;

    // workspace layout (dwords): A0 frags | A1 frags | dy | dx | mask
    unsigned* wsA0 = (unsigned*)d_ws;              // 9*8*2*32*8   = 36864
    unsigned* wsA1 = wsA0 + 36864;                 // 9*8*16*32*8  = 294912
    float* dyA = (float*)(wsA1 + 294912);          // 2*9*4096     = 73728
    float* dxA = dyA + 73728;
    float* mkA = dxA + 73728;

    repack_off<<<144, 256, 0, stream>>>(w_om, wsA0);
    repack_dcn<<<1152, 256, 0, stream>>>(w_dcn, wsA1);
    conv_offset<<<B_ * H_, 256, 0, stream>>>(of, b_om, wsA0, dyA, dxA, mkA);
    dcn_main<<<B_ * H_, 256, 0, stream>>>(x, b_dcn, wsA1, dyA, dxA, mkA, out);
}
